// SingleTemplateEmbedding_23476291239981
// MI455X (gfx1250) — matrix operand order repaired and checked by reading, not compiled, not run on hardware
//
#include <hip/hip_runtime.h>
#include <hip/hip_bf16.h>
#include <cstdint>

// ---------------------------------------------------------------------------
// SingleTemplateEmbedding forward for MI455X (gfx1250, wave32, WMMA).
//
// Precision strategy: activations/weights stay f32 in memory (16MB act tensor
// lives in the 192MB L2); every dense contraction is lowered to
// v_wmma_f32_16x16x32_f16 (f16 inputs converted at the LDS tile boundary,
// f32 accumulate). LN/softmax/sigmoid epilogues run in f32 VALU.
//
// All WMMA GEMM launches are tile-exact (M%64==0, N%64==0, K%32==0) so the
// global->LDS staging is guard-free: float4 global loads -> cvt -> b64 LDS
// stores, and A-fragment reads are contiguous ds_load_b128 (B reads strided per k) per the documented
// CDNA5 A/B VGPR layouts. The N=4 "b2d" projection runs on a scalar kernel.
//
// Workspace requirement: ~135 MB (8 x 16MB f32 buffers + 1MB b2d + small frame data).
// Param pointer order assumes jax pytree flattening (sorted dict keys).
// ---------------------------------------------------------------------------

typedef _Float16 v16h __attribute__((ext_vector_type(16)));
typedef _Float16 h8   __attribute__((ext_vector_type(8)));
typedef _Float16 h4   __attribute__((ext_vector_type(4)));
typedef float    v8f  __attribute__((ext_vector_type(8)));
typedef float    f4   __attribute__((ext_vector_type(4)));

#define NRES 256
#define CDIM 64
#define ROWS (NRES * NRES)            // 65536
#define ACTN ((size_t)ROWS * CDIM)    // 4M floats = 16MB

__device__ __forceinline__ float sigf(float x) { return 1.0f / (1.0f + __expf(-x)); }

__device__ __forceinline__ v8f wmma_f16(v16h a, v16h b, v8f c) {
  return __builtin_amdgcn_wmma_f32_16x16x32_f16(false, a, false, b, (short)0, c, false, false);
}

// A-fragment (16x32 f16) from LDS, documented layout:
// lane L: M = L&15; lanes<16 -> K 0..7 & 16..23, lanes>=16 -> K 8..15 & 24..31.
__device__ __forceinline__ v16h fragA_lds(const _Float16* base, int ld, int m, int lane) {
  int kb = (lane < 16) ? 0 : 8;
  const _Float16* p = base + m * ld + kb;
  h8 lo = *(const h8*)(p);
  h8 hi = *(const h8*)(p + 16);
  v16h a;
#pragma unroll
  for (int i = 0; i < 8; ++i) { a[i] = lo[i]; a[i + 8] = hi[i]; }
  return a;
}

// B-fragment (32x16 f16): lane L: N = L&15; lanes<16 -> K 0..7 & 16..23, lanes>=16 -> K 8..15 & 24..31.
__device__ __forceinline__ v16h fragB_lds(const _Float16* bp, int ld, int lane) {
  int kb = (lane < 16) ? 0 : 8;
  v16h b;
#pragma unroll
  for (int i = 0; i < 8; ++i) { b[i] = bp[(kb + i) * ld]; b[i + 8] = bp[(kb + 16 + i) * ld]; }
  return b;
}

// float4 global -> f16x4 LDS
__device__ __forceinline__ void cvt_store4(_Float16* dst, const float* src) {
  f4 v = *(const f4*)src;
  h4 hv;
#pragma unroll
  for (int i = 0; i < 4; ++i) hv[i] = (_Float16)v[i];
  *(h4*)dst = hv;
}

// ---------------------------------------------------------------------------
// LayerNorm over last dim (64). One wave per (i,j) row.
// ---------------------------------------------------------------------------
__global__ void ln_k(const float* __restrict__ in, const float* __restrict__ s,
                     const float* __restrict__ b, float* __restrict__ out) {
  int row = blockIdx.x;
  int lane = threadIdx.x;
  const float* p = in + (size_t)row * CDIM;
  float a0 = p[lane], a1 = p[lane + 32];
  float sum = a0 + a1;
#pragma unroll
  for (int off = 16; off > 0; off >>= 1) sum += __shfl_xor(sum, off, 32);
  float mean = sum * (1.0f / 64.0f);
  float d0 = a0 - mean, d1 = a1 - mean;
  float v = d0 * d0 + d1 * d1;
#pragma unroll
  for (int off = 16; off > 0; off >>= 1) v += __shfl_xor(v, off, 32);
  float inv = rsqrtf(v * (1.0f / 64.0f) + 1e-5f);
  out[(size_t)row * CDIM + lane]      = d0 * inv * s[lane]      + b[lane];
  out[(size_t)row * CDIM + lane + 32] = d1 * inv * s[lane + 32] + b[lane + 32];
}

// ---------------------------------------------------------------------------
// Generic WMMA GEMM: out = epilogue(alpha * A[MxK] @ B[KxN]).
// 128 threads = 4 waves; 64x64 tile per block; K stepped by 32.
// Requires M%64==0, N%64==0, K%32==0 (all call sites satisfy this).
// epi: 0 store, 1 +bias, 2 relu(+bias), 3 mask*aux*sigmoid(+bias),
//      4 res + (+bias), 5 res + aux*sigmoid(+bias)
// ---------------------------------------------------------------------------
__global__ void __launch_bounds__(128)
gemm_k(const float* __restrict__ A, const float* __restrict__ B,
       const float* __restrict__ bias, const float* __restrict__ aux,
       const float* __restrict__ mrow, const float* __restrict__ res,
       float* __restrict__ out,
       int M, int N, int K, float alpha, int epi) {
  __shared__ __align__(16) _Float16 Al[64][40];
  __shared__ __align__(16) _Float16 Bl[32][72];
  int t = threadIdx.x, w = t >> 5, lane = t & 31;
  int mB = blockIdx.x * 64, nB = blockIdx.y * 64;
  v8f acc[4] = {};
  for (int k0 = 0; k0 < K; k0 += 32) {
#pragma unroll
    for (int it = 0; it < 4; ++it) {           // A tile: 64x32, float4 loads
      int vi = it * 128 + t;
      int r = vi >> 3, c4 = (vi & 7) * 4;
      cvt_store4(&Al[r][c4], &A[(size_t)(mB + r) * K + k0 + c4]);
    }
#pragma unroll
    for (int it = 0; it < 4; ++it) {           // B tile: 32x64, float4 loads
      int vi = it * 128 + t;
      int r = vi >> 4, c4 = (vi & 15) * 4;
      cvt_store4(&Bl[r][c4], &B[(size_t)(k0 + r) * N + nB + c4]);
    }
    __syncthreads();
    v16h a = fragA_lds(&Al[0][0], 40, w * 16 + (lane & 15), lane);
#pragma unroll
    for (int nt = 0; nt < 4; ++nt) {
      v16h bb = fragB_lds(&Bl[0][nt * 16 + (lane & 15)], 72, lane);
      acc[nt] = wmma_f16(a, bb, acc[nt]);
    }
    __syncthreads();
  }
#pragma unroll
  for (int nt = 0; nt < 4; ++nt) {
#pragma unroll
    for (int r = 0; r < 8; ++r) {
      int m = mB + w * 16 + r + ((lane < 16) ? 0 : 8);
      int n = nB + nt * 16 + (lane & 15);
      float v = alpha * acc[nt][r];
      size_t o = (size_t)m * N + n;
      float val;
      if      (epi == 0) val = v;
      else if (epi == 1) val = v + bias[n];
      else if (epi == 2) val = fmaxf(v + bias[n], 0.f);
      else if (epi == 3) val = mrow[m] * aux[o] * sigf(v + bias[n]);
      else if (epi == 4) val = res[o] + v + bias[n];
      else               val = res[o] + aux[o] * sigf(v + bias[n]);
      out[o] = val;
    }
  }
}

// ---------------------------------------------------------------------------
// Per-channel triangle-multiply GEMM. Operands are pre-shuffled channel-major:
//   Ac: [64][256(M)][256(K)] row-major, Bc: [64][256(K)][256(N)] row-major,
// Out_c = A_c @ B_c, written back to the [i,j,c] activation layout.
// ---------------------------------------------------------------------------
__global__ void __launch_bounds__(128)
chan_gemm_k(const float* __restrict__ Ac, const float* __restrict__ Bc,
            float* __restrict__ out) {
  __shared__ __align__(16) _Float16 Al[64][40];
  __shared__ __align__(16) _Float16 Bl[32][72];
  int t = threadIdx.x, w = t >> 5, lane = t & 31;
  int iB = blockIdx.x * 64, jB = blockIdx.y * 64, c = blockIdx.z;
  const float* A = Ac + (size_t)c * ROWS;
  const float* B = Bc + (size_t)c * ROWS;
  v8f acc[4] = {};
  for (int k0 = 0; k0 < NRES; k0 += 32) {
#pragma unroll
    for (int it = 0; it < 4; ++it) {
      int vi = it * 128 + t;
      int r = vi >> 3, c4 = (vi & 7) * 4;
      cvt_store4(&Al[r][c4], &A[(size_t)(iB + r) * NRES + k0 + c4]);
    }
#pragma unroll
    for (int it = 0; it < 4; ++it) {
      int vi = it * 128 + t;
      int r = vi >> 4, c4 = (vi & 15) * 4;
      cvt_store4(&Bl[r][c4], &B[(size_t)(k0 + r) * NRES + jB + c4]);
    }
    __syncthreads();
    v16h a = fragA_lds(&Al[0][0], 40, w * 16 + (lane & 15), lane);
#pragma unroll
    for (int nt = 0; nt < 4; ++nt) {
      v16h bb = fragB_lds(&Bl[0][nt * 16 + (lane & 15)], 72, lane);
      acc[nt] = wmma_f16(a, bb, acc[nt]);
    }
    __syncthreads();
  }
#pragma unroll
  for (int nt = 0; nt < 4; ++nt) {
#pragma unroll
    for (int r = 0; r < 8; ++r) {
      int i = iB + w * 16 + r + ((lane < 16) ? 0 : 8);
      int j = jB + nt * 16 + (lane & 15);
      out[((size_t)i * NRES + j) * CDIM + c] = acc[nt][r];
    }
  }
}

// ---------------------------------------------------------------------------
// b2d projection (N=4): B2D[(q*256+k)*4+h] = sum_c X[q,k,c]*w[c,h].
// Tiny (17M MACs) -> scalar kernel, keeps the WMMA GEMM guard-free.
// ---------------------------------------------------------------------------
__global__ void b2d_k(const float* __restrict__ X, const float* __restrict__ w,
                      float* __restrict__ out) {
  int t = blockIdx.x * 256 + threadIdx.x;   // ROWS*4 threads
  int row = t >> 2, h = t & 3;
  const float* xp = X + (size_t)row * CDIM;
  float acc = 0.f;
#pragma unroll
  for (int c = 0; c < CDIM; ++c) acc += xp[c] * w[c * 4 + h];
  out[t] = acc;
}

// ---------------------------------------------------------------------------
// Triangle attention core: one wave per (qtile16, head, batch-row).
// logits = Q Ktile^T (WMMA, K=16 zero-padded to 32) + mask bias + b2d,
// softmax over 256 in LDS, then S @ V (WMMA over K=256), fused sigmoid gating.
// ---------------------------------------------------------------------------
__global__ void attn_k(const float* __restrict__ Q, const float* __restrict__ Kt,
                       const float* __restrict__ V, const float* __restrict__ G,
                       const float* __restrict__ B2D, const float* __restrict__ gb,
                       const float* __restrict__ mask, int mask_t,
                       float* __restrict__ O) {
  __shared__ __align__(16) float     S[16][264];
  __shared__ __align__(16) _Float16  S16[16][264];
  int lane = threadIdx.x;
  int qt = blockIdx.x, h = blockIdx.y, b = blockIdx.z;
  int h16 = h * 16;
  int m = lane & 15;
  int kb = (lane < 16) ? 0 : 8;
  int hi8 = (lane < 16) ? 0 : 8;
  int qlane = qt * 16 + m;

  // A fragment from Q (M=q rows, contraction dim d=16, upper half zero-padded)
  v16h aq;
  {
    const float* qp = Q + ((size_t)(b * NRES + qlane)) * CDIM + h16 + kb;
#pragma unroll
    for (int e = 0; e < 8; ++e) aq[e] = (_Float16)qp[e];
#pragma unroll
    for (int e = 8; e < 16; ++e) aq[e] = (_Float16)0.f;
  }

  for (int kt = 0; kt < 16; ++kt) {
    v16h bk;   // B fragment: lane&15 = key column, elements = d (kb..kb+7), upper half zero
    {
#pragma unroll
      for (int e = 0; e < 8; ++e)
        bk[e] = (_Float16)Kt[((size_t)(b * NRES + kt * 16 + m)) * CDIM + h16 + kb + e];
#pragma unroll
      for (int e = 8; e < 16; ++e) bk[e] = (_Float16)0.f;
    }
    v8f c = {};
    c = wmma_f16(aq, bk, c);
    int n = lane & 15;
    int kg = kt * 16 + n;
    size_t mi = mask_t ? ((size_t)kg * NRES + b) : ((size_t)b * NRES + kg);
    float mb = 1e9f * (mask[mi] - 1.f);
#pragma unroll
    for (int r = 0; r < 8; ++r) {
      int qg = qt * 16 + r + hi8;
      S[r + hi8][kg] = c[r] + mb + B2D[((size_t)qg * NRES + kg) * 4 + h];
    }
  }
  __syncthreads();

  { // softmax: 2 lanes per row, 128 columns each
    int rr = lane >> 1, j0 = (lane & 1) * 128;
    float mx = -1e30f;
    for (int j = 0; j < 128; ++j) mx = fmaxf(mx, S[rr][j0 + j]);
    mx = fmaxf(mx, __shfl_xor(mx, 1, 32));
    float sum = 0.f;
    for (int j = 0; j < 128; ++j) sum += __expf(S[rr][j0 + j] - mx);
    sum += __shfl_xor(sum, 1, 32);
    float inv = 1.f / sum;
    for (int j = 0; j < 128; ++j)
      S16[rr][j0 + j] = (_Float16)(__expf(S[rr][j0 + j] - mx) * inv);
  }
  __syncthreads();

  v8f o = {};
  for (int kk = 0; kk < 8; ++kk) {
    v16h as = fragA_lds(&S16[0][kk * 32], 264, m, lane);
    v16h bv;  // lane&15 = d, elements = key index (table order)
    int kp = kk * 32 + kb;
#pragma unroll
    for (int e = 0; e < 8; ++e) {
      bv[e] = (_Float16)V[((size_t)(b * NRES + kp + e)) * CDIM + h16 + m];
      bv[e + 8] = (_Float16)V[((size_t)(b * NRES + kp + 16 + e)) * CDIM + h16 + m];
    }
    o = wmma_f16(as, bv, o);
  }
  int d = lane & 15;
#pragma unroll
  for (int r = 0; r < 8; ++r) {
    int qg = qt * 16 + r + hi8;
    size_t idx = ((size_t)(b * NRES + qg)) * CDIM + h16 + d;
    float g = G[idx] + gb[h16 + d];
    O[idx] = o[r] * sigf(g);
  }
}

// ---------------------------------------------------------------------------
// Layout shuffles
// ---------------------------------------------------------------------------
__global__ void transpose_nnc_k(const float* __restrict__ src, float* __restrict__ dst) {
  size_t t = (size_t)blockIdx.x * 256 + threadIdx.x;
  int c = (int)(t & 63);
  size_t ij = t >> 6;
  int j = (int)(ij & 255), i = (int)(ij >> 8);
  dst[(((size_t)j * NRES) + i) * CDIM + c] = src[t];
}

__global__ void nnc_to_cnn_k(const float* __restrict__ src, float* __restrict__ dst, int swap) {
  size_t t = (size_t)blockIdx.x * 256 + threadIdx.x;   // dst: c*65536 + a*256 + b
  int bcol = (int)(t & 255);
  int a = (int)((t >> 8) & 255);
  int c = (int)(t >> 16);
  size_t sidx = ((size_t)(swap ? (bcol * NRES + a) : (a * NRES + bcol))) * CDIM + c;
  dst[t] = src[sidx];
}

// ---------------------------------------------------------------------------
// Per-residue frames (e1,e2,e3) + backbone mask
// ---------------------------------------------------------------------------
__global__ void frames_k(const float* __restrict__ pos, const float* __restrict__ am,
                         float* __restrict__ rot, float* __restrict__ bbm) {
  int i = threadIdx.x;
  const float* p = pos + (size_t)i * 111;    // 37 atoms * 3
  float nx = p[0], ny = p[1], nz = p[2];
  float cax = p[3], cay = p[4], caz = p[5];
  float cx = p[6], cy = p[7], cz = p[8];
  float e1x = cx - cax, e1y = cy - cay, e1z = cz - caz;
  float inv = rsqrtf(e1x * e1x + e1y * e1y + e1z * e1z + 1e-12f);
  e1x *= inv; e1y *= inv; e1z *= inv;
  float v2x = nx - cax, v2y = ny - cay, v2z = nz - caz;
  float d = e1x * v2x + e1y * v2y + e1z * v2z;
  float e2x = v2x - e1x * d, e2y = v2y - e1y * d, e2z = v2z - e1z * d;
  inv = rsqrtf(e2x * e2x + e2y * e2y + e2z * e2z + 1e-12f);
  e2x *= inv; e2y *= inv; e2z *= inv;
  float e3x = e1y * e2z - e1z * e2y;
  float e3y = e1z * e2x - e1x * e2z;
  float e3z = e1x * e2y - e1y * e2x;
  float* r = rot + i * 9;
  r[0] = e1x; r[1] = e1y; r[2] = e1z;
  r[3] = e2x; r[4] = e2y; r[5] = e2z;
  r[6] = e3x; r[7] = e3y; r[8] = e3z;
  bbm[i] = am[i * 37 + 0] * am[i * 37 + 1] * am[i * 37 + 2];
}

// ---------------------------------------------------------------------------
// Featurize (88 feats in LDS) + embed (88->64) per (i,j)
// ---------------------------------------------------------------------------
__global__ void feat_embed_k(const float* __restrict__ pb, const float* __restrict__ pbm,
                             const int* __restrict__ aat, const float* __restrict__ pos,
                             const float* __restrict__ rot, const float* __restrict__ bbm,
                             const float* __restrict__ ew, const float* __restrict__ eb,
                             float* __restrict__ act) {
  __shared__ float f[88];
  int bx = blockIdx.x;
  int i = bx >> 8, j = bx & 255;
  int t = threadIdx.x;   // 128
  float bb2 = bbm[i] * bbm[j];
  if (t < 88) {
    float val = 0.f;
    if (t < 39) {
      float dx = pb[i * 3 + 0] - pb[j * 3 + 0];
      float dy = pb[i * 3 + 1] - pb[j * 3 + 1];
      float dz = pb[i * 3 + 2] - pb[j * 3 + 2];
      float d2 = dx * dx + dy * dy + dz * dz;
      float lo = 3.25f + 1.25f * t; lo = lo * lo;
      float hh = 3.25f + 1.25f * (t + 1);
      float hi = (t < 38) ? hh * hh : 1e8f;
      val = (d2 > lo && d2 < hi) ? 1.f : 0.f;
    } else if (t == 39) {
      val = pbm[i] * pbm[j];
    } else if (t < 62) {
      val = (aat[j] == (t - 40)) ? 1.f : 0.f;
    } else if (t < 84) {
      val = (aat[i] == (t - 62)) ? 1.f : 0.f;
    } else if (t < 87) {
      float rx = pos[(size_t)j * 111 + 3] - pos[(size_t)i * 111 + 3];
      float ry = pos[(size_t)j * 111 + 4] - pos[(size_t)i * 111 + 4];
      float rz = pos[(size_t)j * 111 + 5] - pos[(size_t)i * 111 + 5];
      const float* r = rot + i * 9;
      float v0 = r[0] * rx + r[1] * ry + r[2] * rz;
      float v1 = r[3] * rx + r[4] * ry + r[5] * rz;
      float v2 = r[6] * rx + r[7] * ry + r[8] * rz;
      float invd = rsqrtf(1e-6f + v0 * v0 + v1 * v1 + v2 * v2) * bb2;
      float vv = (t == 84) ? v0 : ((t == 85) ? v1 : v2);
      val = vv * invd;
    } else {
      val = bb2;
    }
    f[t] = val * bb2;
  }
  __syncthreads();
  if (t < 64) {
    float acc = eb[t];
    for (int u = 0; u < 88; ++u) acc += f[u] * ew[u * 64 + t];
    act[(size_t)bx * 64 + t] = acc;
  }
}

// ---------------------------------------------------------------------------
// Host orchestration
// ---------------------------------------------------------------------------
extern "C" void kernel_launch(void* const* d_in, const int* in_sizes, int n_in,
                              void* d_out, int out_size, void* d_ws, size_t ws_size,
                              hipStream_t stream) {
  (void)in_sizes; (void)n_in; (void)out_size; (void)ws_size;
  const int*   aat  = (const int*)d_in[1];
  const float* pb   = (const float*)d_in[2];
  const float* pbm  = (const float*)d_in[3];
  const float* pos  = (const float*)d_in[4];
  const float* am   = (const float*)d_in[5];
  const float* mask = (const float*)d_in[6];
  auto P = [&](int i) { return (const float*)d_in[i]; };
  // jax pytree (sorted keys): per block (58 leaves starting at 7):
  //  pt:  +0 b1 +1 b2 +2 ln_b +3 ln_s +4 w1 +5 w2
  //  tae: +6 b2d_w +7 g_b +8 g_w +9 k_w +10 ln_b +11 ln_s +12 o_b +13 o_w +14 q_w +15 v_w
  //  tas: +16..+25 (same order)
  //  tmi: +26 cln_b +27 cln_s +28 gl_b +29 gl_w +30 lg_b +31 lg_w +32 ln_b +33 ln_s
  //       +34 lp_b +35 lp_w +36 op_b +37 op_w +38 rg_b +39 rg_w +40 rp_b +41 rp_w
  //  tmo: +42..+57
  // then 123 embed_b, 124 embed_w, 125 out_ln_b, 126 out_ln_s

  float* W = (float*)d_ws;
  const size_t M4 = ACTN;
  float* ACT = W;
  float* X   = W + 1 * M4;
  float* Bq  = W + 2 * M4;
  float* Bk  = W + 3 * M4;
  float* Bv  = W + 4 * M4;
  float* Bg  = W + 5 * M4;
  float* B5  = W + 6 * M4;
  float* T0  = W + 7 * M4;
  float* B2D = W + 8 * M4;                  // 262144 floats
  float* ROT = B2D + (size_t)ROWS * 4;      // 2304 floats
  float* BBM = ROT + 256 * 9;               // 256 floats

  auto gemm = [&](const float* A, const float* Bw, const float* bias, const float* aux,
                  const float* mrow, const float* res, float* out,
                  int M, int N, int K, float alpha, int epi) {
    dim3 g(M / 64, N / 64);
    gemm_k<<<g, dim3(128), 0, stream>>>(A, Bw, bias, aux, mrow, res, out, M, N, K, alpha, epi);
  };
  auto lnl = [&](const float* in, const float* s, const float* b, float* out) {
    ln_k<<<dim3(ROWS), dim3(32), 0, stream>>>(in, s, b, out);
  };

  frames_k<<<dim3(1), dim3(256), 0, stream>>>(pos, am, ROT, BBM);
  feat_embed_k<<<dim3(ROWS), dim3(128), 0, stream>>>(pb, pbm, aat, pos, ROT, BBM,
                                                     P(124), P(123), ACT);

  auto run_attn = [&](float* src, int ab, int mask_t) {
    lnl(src, P(ab + 11), P(ab + 10), X);
    gemm(X, P(ab + 14), nullptr, nullptr, nullptr, nullptr, Bq, ROWS, 64, 64, 0.25f, 0); // q (scaled)
    gemm(X, P(ab + 9),  nullptr, nullptr, nullptr, nullptr, Bk, ROWS, 64, 64, 1.f,   0); // k
    gemm(X, P(ab + 15), nullptr, nullptr, nullptr, nullptr, Bv, ROWS, 64, 64, 1.f,   0); // v
    gemm(X, P(ab + 8),  nullptr, nullptr, nullptr, nullptr, Bg, ROWS, 64, 64, 1.f,   0); // gate proj
    b2d_k<<<dim3(ROWS / 64), dim3(256), 0, stream>>>(X, P(ab + 6), B2D);                 // b2d (N=4)
    attn_k<<<dim3(16, 4, NRES), dim3(32), 0, stream>>>(Bq, Bk, Bv, Bg, B2D, P(ab + 7),
                                                       mask, mask_t, B5);
    gemm(B5, P(ab + 13), P(ab + 12), nullptr, nullptr, src, src, ROWS, 64, 64, 1.f, 4); // out proj + residual
  };

  auto run_mult = [&](int mb, bool outgoing) {
    lnl(ACT, P(mb + 7), P(mb + 6), X);
    gemm(X, P(mb + 9),  P(mb + 8),  nullptr, nullptr, nullptr, Bq, ROWS, 64, 64, 1.f, 1); // left proj
    gemm(X, P(mb + 5),  P(mb + 4),  Bq, mask, nullptr,        Bq, ROWS, 64, 64, 1.f, 3); // left gated
    gemm(X, P(mb + 15), P(mb + 14), nullptr, nullptr, nullptr, Bk, ROWS, 64, 64, 1.f, 1); // right proj
    gemm(X, P(mb + 13), P(mb + 12), Bk, mask, nullptr,        Bk, ROWS, 64, 64, 1.f, 3); // right gated
    if (outgoing) {  // out[i,j,c] = sum_k L[i,k,c] R[j,k,c]  => A[c][i][k], B[c][k][j]
      nnc_to_cnn_k<<<dim3(16384), dim3(256), 0, stream>>>(Bq, Bv, 0);  // A = L[i,k,c]
      nnc_to_cnn_k<<<dim3(16384), dim3(256), 0, stream>>>(Bk, Bg, 1);  // B[k][j] = R[j,k,c]
    } else {         // out[i,j,c] = sum_k R[k,i,c] L[k,j,c]  => A[c][i][k], B[c][k][j]
      nnc_to_cnn_k<<<dim3(16384), dim3(256), 0, stream>>>(Bk, Bv, 1);  // A[i][k] = R[k,i,c]
      nnc_to_cnn_k<<<dim3(16384), dim3(256), 0, stream>>>(Bq, Bg, 0);  // B[k][j] = L[k,j,c]
    }
    chan_gemm_k<<<dim3(4, 4, 64), dim3(128), 0, stream>>>(Bv, Bg, B5);
    lnl(B5, P(mb + 1), P(mb + 0), T0);                                                     // center LN
    gemm(T0, P(mb + 11), P(mb + 10), nullptr, nullptr, nullptr, Bq, ROWS, 64, 64, 1.f, 1); // out proj
    gemm(X,  P(mb + 3),  P(mb + 2),  Bq, nullptr, ACT, ACT,      ROWS, 64, 64, 1.f, 5);    // gate + residual
  };

  for (int blk = 0; blk < 2; ++blk) {
    int base = 7 + blk * 58;
    int pt = base + 0, tae = base + 6, tas = base + 16, tmi = base + 26, tmo = base + 42;

    run_attn(ACT, tas, 0);                                                  // row attention
    transpose_nnc_k<<<dim3(16384), dim3(256), 0, stream>>>(ACT, T0);        // column attention
    run_attn(T0, tae, 1);
    transpose_nnc_k<<<dim3(16384), dim3(256), 0, stream>>>(T0, ACT);
    run_mult(tmo, true);                                                    // triangle outgoing
    run_mult(tmi, false);                                                   // triangle incoming

    // transition, 4 row-chunks of 16384 (hidden 16384x256 fits B5)
    lnl(ACT, P(pt + 3), P(pt + 2), X);
    for (int ch = 0; ch < 4; ++ch) {
      size_t off = (size_t)ch * 16384 * 64;
      gemm(X + off, P(pt + 4), P(pt + 0), nullptr, nullptr, nullptr, B5,
           16384, 256, 64, 1.f, 2);                                          // relu(x@w1+b1)
      gemm(B5, P(pt + 5), P(pt + 1), nullptr, nullptr, ACT + off, ACT + off,
           16384, 64, 256, 1.f, 4);                                          // act += h@w2+b2
    }
  }

  lnl(ACT, P(126), P(125), (float*)d_out);   // final output LN
}
